// MyLSTM_33552284516751
// MI455X (gfx1250) — matrix operand order repaired and checked by reading, not compiled, not run on hardware
//
#include <hip/hip_runtime.h>

// ---------------------------------------------------------------------------
// CDNA5 (gfx1250) LSTM.  wave32, V_WMMA_F32_16X16X32_BF16 with hi/lo bf16
// split (3 WMMAs per K=32 tile) for ~fp32 accuracy at matrix-core speed.
// ---------------------------------------------------------------------------

typedef __attribute__((ext_vector_type(16))) __bf16 v16bf;
typedef __attribute__((ext_vector_type(8)))  __bf16 v8bf;
typedef __attribute__((ext_vector_type(8)))  float  v8f;

constexpr int kI  = 512;   // INPUT_SIZE
constexpr int kH  = 1024;  // HIDDEN_SIZE
constexpr int kB  = 64;    // BATCH
constexpr int kT  = 512;   // SEQ
constexpr int k4H = 4 * kH;

static __device__ __forceinline__ float sigmoid_fast(float x) {
  return 1.0f / (1.0f + __expf(-x));
}

static __device__ __forceinline__ v8f wmma_bf16(v16bf a, v16bf b, v8f c) {
  // (neg_a, A, neg_b, B, c_mod, C, reuse_a, reuse_b)
  return __builtin_amdgcn_wmma_f32_16x16x32_bf16(false, a, false, b, (short)0,
                                                 c, false, false);
}

// B fragment: lane holds column n = lane&15, 16 contiguous K halves starting
// at K = kt + (lane>>4)*16  (weights stored transposed, [N][K] row-major).
static __device__ __forceinline__ v16bf load_b16(const __bf16* p) {
  union { v16bf v; v8bf h[2]; } u;
  u.h[0] = *(const v8bf*)(p);
  u.h[1] = *(const v8bf*)(p + 8);
  return u.v;
}

// ---------------------------------------------------------------------------
// Repack one gate-concatenated weight matrix W (4 gate blocks, each [K][kH]
// row-major) into transposed bf16 hi/lo arrays Wt[4H][K].
// ---------------------------------------------------------------------------
__global__ void wt_repack_kernel(const float* __restrict__ g0,
                                 const float* __restrict__ g1,
                                 const float* __restrict__ g2,
                                 const float* __restrict__ g3, int K,
                                 __bf16* __restrict__ hi,
                                 __bf16* __restrict__ lo) {
  const size_t idx = (size_t)blockIdx.x * blockDim.x + threadIdx.x;
  const size_t total = (size_t)k4H * (size_t)K;
  if (idx >= total) return;
  const int n = (int)(idx / K);
  const int k = (int)(idx % K);
  const float* src = (n < kH) ? g0 : (n < 2 * kH) ? g1 : (n < 3 * kH) ? g2 : g3;
  const float v = src[(size_t)k * kH + (n & (kH - 1))];
  const __bf16 h = (__bf16)v;
  hi[idx] = h;
  lo[idx] = (__bf16)(v - (float)h);
}

__global__ void state_init_kernel(__bf16* __restrict__ h_hi,
                                  __bf16* __restrict__ h_lo,
                                  float* __restrict__ c) {
  const int i = blockIdx.x * blockDim.x + threadIdx.x;
  if (i < kB * kH) {
    h_hi[i] = (__bf16)0.0f;
    h_lo[i] = (__bf16)0.0f;
    c[i] = 0.0f;
  }
}

// ---------------------------------------------------------------------------
// Phase 1: xU[t][b][4H] = x[b][t][:] @ Ut^T + bias.   M = T*B rows (r=t*64+b),
// N = 4096, K = 512.  Block = 4 waves = 64 rows x 64 cols (A reused across
// 4 N-tiles).  x is converted to bf16 hi/lo in registers.
// ---------------------------------------------------------------------------
__global__ __launch_bounds__(128) void xu_gemm_kernel(
    const float* __restrict__ x, const __bf16* __restrict__ Ut_hi,
    const __bf16* __restrict__ Ut_lo, const float* __restrict__ b_i,
    const float* __restrict__ b_f, const float* __restrict__ b_c,
    const float* __restrict__ b_o, float* __restrict__ xU) {
  const int lane = threadIdx.x & 31;
  const int wave = threadIdx.x >> 5;
  const int lr = lane & 15;
  const int hs = lane >> 4;
  const int r0 = blockIdx.x * 64 + wave * 16;
  const int n0 = blockIdx.y * 64;

  // A row for this lane: global row r = t*64+b ; x is [B][T][I]
  const int rA = r0 + lr;
  const int tA = rA >> 6;
  const int bA = rA & 63;
  const float* arow = x + ((size_t)bA * kT + tA) * kI + hs * 16;

  v8f acc[4] = {};
  for (int kt = 0; kt < kI; kt += 32) {
    v16bf ahi, alo;
#pragma unroll
    for (int i = 0; i < 8; ++i) {
      const float v0 = arow[kt + i];
      const float v1 = arow[kt + 8 + i];
      const __bf16 h0 = (__bf16)v0;
      const __bf16 h1 = (__bf16)v1;
      ahi[i] = h0;
      alo[i] = (__bf16)(v0 - (float)h0);
      ahi[i + 8] = h1;
      alo[i + 8] = (__bf16)(v1 - (float)h1);
    }
#pragma unroll
    for (int nt = 0; nt < 4; ++nt) {
      const size_t boff = (size_t)(n0 + nt * 16 + lr) * kI + hs * 16 + kt;
      const v16bf bhi = load_b16(Ut_hi + boff);
      const v16bf blo = load_b16(Ut_lo + boff);
      acc[nt] = wmma_bf16(ahi, bhi, acc[nt]);
      acc[nt] = wmma_bf16(ahi, blo, acc[nt]);
      acc[nt] = wmma_bf16(alo, bhi, acc[nt]);
    }
  }

  const float* bias[4] = {b_i, b_f, b_c, b_o};
#pragma unroll
  for (int nt = 0; nt < 4; ++nt) {
    const int col = n0 + nt * 16 + lr;
    const float bv = bias[col >> 10][col & (kH - 1)];
#pragma unroll
    for (int j = 0; j < 8; ++j) {
      const int r = r0 + hs * 8 + j;  // D: lane<16 -> rows 0..7, else 8..15
      xU[(size_t)r * k4H + col] = acc[nt][j] + bv;
    }
  }
}

// ---------------------------------------------------------------------------
// Phase 2: one fused timestep.  64 blocks x 128 threads; block owns H-columns
// [16*blockIdx.x, +16) of all four gates; wave w owns batch rows [16w, +16).
// gates = xu_t + h @ V (bf16x3 WMMA), then sigmoid/sigmoid cell update,
// writes h to output and re-splits h into bf16 hi/lo for the next step.
// ---------------------------------------------------------------------------
__global__ __launch_bounds__(128) void lstm_step_kernel(
    const float* __restrict__ xU, const __bf16* __restrict__ Vt_hi,
    const __bf16* __restrict__ Vt_lo, const __bf16* __restrict__ hin_hi,
    const __bf16* __restrict__ hin_lo, __bf16* __restrict__ hout_hi,
    __bf16* __restrict__ hout_lo, float* __restrict__ c_state,
    float* __restrict__ out_h, float* __restrict__ out_hT,
    float* __restrict__ out_cT, int t) {
  const int lane = threadIdx.x & 31;
  const int wave = threadIdx.x >> 5;
  const int lr = lane & 15;
  const int hs = lane >> 4;
  const int n0 = blockIdx.x << 4;
  const int m0 = wave << 4;

  const __bf16* arow_hi = hin_hi + (size_t)(m0 + lr) * kH + hs * 16;
  const __bf16* arow_lo = hin_lo + (size_t)(m0 + lr) * kH + hs * 16;
  const size_t brow = (size_t)(n0 + lr) * kH + hs * 16;

  v8f acc[4] = {};
  for (int kt = 0; kt < kH; kt += 32) {
    union { v16bf v; v8bf h[2]; } ua;
    ua.h[0] = *(const v8bf*)(arow_hi + kt);
    ua.h[1] = *(const v8bf*)(arow_hi + kt + 8);
    const v16bf ahi = ua.v;
    ua.h[0] = *(const v8bf*)(arow_lo + kt);
    ua.h[1] = *(const v8bf*)(arow_lo + kt + 8);
    const v16bf alo = ua.v;
#pragma unroll
    for (int g = 0; g < 4; ++g) {
      const size_t off = (size_t)g * kH * kH + brow + kt;
      const v16bf bhi = load_b16(Vt_hi + off);
      const v16bf blo = load_b16(Vt_lo + off);
      acc[g] = wmma_bf16(ahi, bhi, acc[g]);
      acc[g] = wmma_bf16(ahi, blo, acc[g]);
      acc[g] = wmma_bf16(alo, bhi, acc[g]);
    }
  }

  const int colH = n0 + lr;
#pragma unroll
  for (int j = 0; j < 8; ++j) {
    const int b = m0 + hs * 8 + j;
    const float* xu = xU + ((size_t)t * kB + b) * k4H + colH;
    const float iv = sigmoid_fast(acc[0][j] + xu[0 * kH]);
    const float fv = sigmoid_fast(acc[1][j] + xu[1 * kH]);
    const float gv = sigmoid_fast(acc[2][j] + xu[2 * kH]);  // ref uses sigmoid
    const float ov = sigmoid_fast(acc[3][j] + xu[3 * kH]);
    const size_t ci = (size_t)b * kH + colH;
    const float cn = fv * c_state[ci] + iv * gv;
    const float hn = ov * tanhf(cn);
    c_state[ci] = cn;
    out_h[((size_t)b * kT + t) * kH + colH] = hn;  // hidden_seq [B][T][H]
    const __bf16 hh = (__bf16)hn;
    hout_hi[ci] = hh;
    hout_lo[ci] = (__bf16)(hn - (float)hh);
    if (t == kT - 1) {
      out_hT[ci] = hn;
      out_cT[ci] = cn;
    }
  }
}

// ---------------------------------------------------------------------------
extern "C" void kernel_launch(void* const* d_in, const int* in_sizes, int n_in,
                              void* d_out, int out_size, void* d_ws,
                              size_t ws_size, hipStream_t stream) {
  const float* x   = (const float*)d_in[0];
  const float* U_i = (const float*)d_in[1];
  const float* V_i = (const float*)d_in[2];
  const float* b_i = (const float*)d_in[3];
  const float* U_f = (const float*)d_in[4];
  const float* V_f = (const float*)d_in[5];
  const float* b_f = (const float*)d_in[6];
  const float* U_c = (const float*)d_in[7];
  const float* V_c = (const float*)d_in[8];
  const float* b_c = (const float*)d_in[9];
  const float* U_o = (const float*)d_in[10];
  const float* V_o = (const float*)d_in[11];
  const float* b_o = (const float*)d_in[12];

  // Workspace carve-up (~563 MB total, all power-of-two aligned chunks).
  char* p = (char*)d_ws;
  float* xU = (float*)p;       p += (size_t)kT * kB * k4H * sizeof(float);
  __bf16* Vt_hi = (__bf16*)p;  p += (size_t)k4H * kH * sizeof(__bf16);
  __bf16* Vt_lo = (__bf16*)p;  p += (size_t)k4H * kH * sizeof(__bf16);
  __bf16* Ut_hi = (__bf16*)p;  p += (size_t)k4H * kI * sizeof(__bf16);
  __bf16* Ut_lo = (__bf16*)p;  p += (size_t)k4H * kI * sizeof(__bf16);
  __bf16* h_hi[2];
  __bf16* h_lo[2];
  h_hi[0] = (__bf16*)p;        p += (size_t)kB * kH * sizeof(__bf16);
  h_hi[1] = (__bf16*)p;        p += (size_t)kB * kH * sizeof(__bf16);
  h_lo[0] = (__bf16*)p;        p += (size_t)kB * kH * sizeof(__bf16);
  h_lo[1] = (__bf16*)p;        p += (size_t)kB * kH * sizeof(__bf16);
  float* c_state = (float*)p;

  float* out_h  = (float*)d_out;                    // [B][T][H]
  float* out_hT = out_h + (size_t)kB * kT * kH;     // [B][H]
  float* out_cT = out_hT + (size_t)kB * kH;         // [B][H]

  // Weight repack (U: K=512, V: K=1024) + zero initial state.
  {
    const int totU = k4H * kI;
    wt_repack_kernel<<<(totU + 255) / 256, 256, 0, stream>>>(
        U_i, U_f, U_c, U_o, kI, Ut_hi, Ut_lo);
    const int totV = k4H * kH;
    wt_repack_kernel<<<(totV + 255) / 256, 256, 0, stream>>>(
        V_i, V_f, V_c, V_o, kH, Vt_hi, Vt_lo);
  }
  state_init_kernel<<<(kB * kH + 255) / 256, 256, 0, stream>>>(h_hi[0], h_lo[0],
                                                               c_state);

  // Phase 1: big parallel GEMM for all timesteps' input projections.
  dim3 gx((kT * kB) / 64, k4H / 64);
  xu_gemm_kernel<<<gx, 128, 0, stream>>>(x, Ut_hi, Ut_lo, b_i, b_f, b_c, b_o,
                                         xU);

  // Phase 2: sequential recurrence, one fused kernel per timestep
  // (kernel boundary = grid-wide sync; h ping-pongs between buffers).
  for (int t = 0; t < kT; ++t) {
    const int rb = t & 1;
    lstm_step_kernel<<<kH / 16, 128, 0, stream>>>(
        xU, Vt_hi, Vt_lo, h_hi[rb], h_lo[rb], h_hi[1 - rb], h_lo[1 - rb],
        c_state, out_h, out_hT, out_cT, t);
  }
}
